// MPNN_13520557048110
// MI455X (gfx1250) — compile-verified, matrix operand order checked by reading, not run on hardware
//
#include <hip/hip_runtime.h>

#define EMB     24
#define NWAVE_  8
#define NANG_   9
#define NORB    32
#define HID     64
#define LDH     68   // LDS row stride in floats (pad vs bank conflicts)

typedef __attribute__((ext_vector_type(16))) _Float16 v16h;
typedef __attribute__((ext_vector_type(8)))  _Float16 v8h;
typedef __attribute__((ext_vector_type(8)))  float    v8f;

__device__ __forceinline__ float silu(float x) {
  // fast: x * v_rcp_f32(1+exp(-x)) -- avoid IEEE div sequence
  return x * __builtin_amdgcn_rcpf(1.0f + __expf(-x));
}

__device__ __forceinline__ v8f wmma_f16(v16h a, v16h b, v8f c) {
  // D(16x16 f32) = A(16x32 f16) * B(32x16 f16) + C
  return __builtin_amdgcn_wmma_f32_16x16x32_f16(false, a, false, b, (short)0, c,
                                                false, false);
}

__device__ __forceinline__ v16h combine16(v8h lo, v8h hi) {
  return __builtin_shufflevector(lo, hi, 0, 1, 2, 3, 4, 5, 6, 7,
                                 8, 9, 10, 11, 12, 13, 14, 15);
}

// ---------------------------------------------------------------------------
// Pre-pack a weight W(K x Nd, row-major f32) into Wt(Npad x K, f16),
// transposed to N-major and zero-padded to Npad columns. After this, a B
// fragment is two contiguous b128 loads per lane, no converts in the hot loop.
// ---------------------------------------------------------------------------
__global__ void pack_w_kernel(const float* __restrict__ W, int K, int Nd, int Npad,
                              _Float16* __restrict__ Wt)
{
  int idx = blockIdx.x * blockDim.x + threadIdx.x;
  if (idx >= Npad * K) return;
  int c = idx / K;
  int k = idx - c * K;
  Wt[idx] = (c < Nd) ? (_Float16)W[(size_t)k * Nd + c] : (_Float16)0.0f;
}

// ---------------------------------------------------------------------------
// Per-atom embedding MLP: species(1) -> 24 -> 24 -> 24   (tiny, VALU)
// ---------------------------------------------------------------------------
__global__ void emb_kernel(const float* __restrict__ species, int n,
                           const float* __restrict__ w0, const float* __restrict__ b0,
                           const float* __restrict__ w1, const float* __restrict__ b1,
                           const float* __restrict__ w2, const float* __restrict__ b2,
                           float* __restrict__ cc)
{
  int i = blockIdx.x * blockDim.x + threadIdx.x;
  if (i >= n) return;
  float s = species[i];
  float h0[EMB], h1[EMB];
  #pragma unroll
  for (int j = 0; j < EMB; ++j) h0[j] = silu(s * w0[j] + b0[j]);
  for (int j = 0; j < EMB; ++j) {
    float acc = b1[j];
    #pragma unroll
    for (int k = 0; k < EMB; ++k) acc += h0[k] * w1[k * EMB + j];
    h1[j] = silu(acc);
  }
  for (int j = 0; j < EMB; ++j) {
    float acc = b2[j];
    #pragma unroll
    for (int k = 0; k < EMB; ++k) acc += h1[k] * w2[k * EMB + j];
    cc[(size_t)i * EMB + j] = acc;
  }
}

// ---------------------------------------------------------------------------
// Per-edge: distvec, cut, radial, sph -> orbital[e][9][8]; co[nl0] += orbital
// ---------------------------------------------------------------------------
__global__ void edge_kernel(const int* __restrict__ nl,
                            const float* __restrict__ cart,
                            const float* __restrict__ shifts,
                            const float* __restrict__ neigh_factor,
                            const float* __restrict__ cc, int E,
                            float* __restrict__ orbital,
                            float* __restrict__ cutbuf,
                            float* __restrict__ co)
{
  int e = blockIdx.x * blockDim.x + threadIdx.x;
  if (e >= E) return;
  int i = nl[e];
  int j = nl[E + e];
  float dx = cart[j*3+0] - cart[i*3+0] + shifts[(size_t)e*3+0];
  float dy = cart[j*3+1] - cart[i*3+1] + shifts[(size_t)e*3+1];
  float dz = cart[j*3+2] - cart[i*3+2] + shifts[(size_t)e*3+2];
  float dist = sqrtf(dx*dx + dy*dy + dz*dz);

  float ne[3 * NWAVE_];
  #pragma unroll
  for (int w = 0; w < 3 * NWAVE_; ++w)
    ne[w] = cc[(size_t)i * EMB + w] * cc[(size_t)j * EMB + w];

  float cv = 0.5f * __cosf(dist * 0.78539816339744831f) + 0.5f; // pi/CUTOFF
  float cut = neigh_factor[e] * cv * cv;
  cutbuf[e] = cut;

  float rad[NWAVE_];
  #pragma unroll
  for (int w = 0; w < NWAVE_; ++w) {
    float d = ne[NWAVE_ + w] * (dist - ne[2 * NWAVE_ + w]);
    rad[w] = __expf(-d * d);
  }

  float ux = dx * 0.25f, uy = dy * 0.25f, uz = dz * 0.25f;  // 1/CUTOFF
  float r2 = ux*ux + uy*uy + uz*uz;
  float sph[NANG_];
  sph[0] = 0.28209479177387814f;
  sph[1] = 0.4886025119029199f * uy;
  sph[2] = 0.4886025119029199f * uz;
  sph[3] = 0.4886025119029199f * ux;
  sph[4] = 1.0925484305920792f * ux * uy;
  sph[5] = 1.0925484305920792f * uy * uz;
  sph[6] = 0.31539156525252005f * (3.0f * uz * uz - r2);
  sph[7] = 1.0925484305920792f * ux * uz;
  sph[8] = 0.5462742152960396f * (ux * ux - uy * uy);

  float* orb = orbital + (size_t)e * 72;
  float* coi = co + (size_t)i * 72;
  #pragma unroll
  for (int a = 0; a < NANG_; ++a) {
    float sa = cut * sph[a];
    #pragma unroll
    for (int w = 0; w < NWAVE_; ++w) {
      float v = sa * rad[w] * ne[w];
      orb[a * NWAVE_ + w] = v;
      atomicAdd(&coi[a * NWAVE_ + w], v);
    }
  }
}

// ---------------------------------------------------------------------------
// density[n][m] (+)= sum_a ( sum_w co[n][a][w] * contracted[l(a)][w][m] )^2
// ---------------------------------------------------------------------------
__global__ void contract_kernel(const float* __restrict__ co,
                                const float* __restrict__ contracted,
                                float* __restrict__ density, int n, int accumulate)
{
  int idx = blockIdx.x * blockDim.x + threadIdx.x;
  if (idx >= n * NORB) return;
  int atom = idx >> 5;
  int m = idx & 31;
  const float* coa = co + (size_t)atom * 72;
  float acc = 0.0f;
  #pragma unroll
  for (int a = 0; a < NANG_; ++a) {
    int l = (a == 0) ? 0 : ((a < 4) ? 1 : 2);
    const float* cm = contracted + (size_t)l * (NWAVE_ * NORB) + m;
    float s = 0.0f;
    #pragma unroll
    for (int w = 0; w < NWAVE_; ++w) s += coa[a * NWAVE_ + w] * cm[(size_t)w * NORB];
    acc += s * s;
  }
  if (accumulate) density[idx] += acc; else density[idx] = acc;
}

// ---------------------------------------------------------------------------
// WMMA MLP: X(n x 32) -> silu(64) -> silu(64) -> out_dim  (out_dim <= 16)
// One wave per 16-row tile. Weights are pre-packed N-major f16 (Npad x K),
// so each B fragment is two contiguous b128 loads. LDS inter-layer transpose
// is private per wave (same-wave DS in-order; s_wait_dscnt fences).
// ---------------------------------------------------------------------------
__global__ __launch_bounds__(128) void mlp_kernel(
    const float* __restrict__ X, int n,
    const _Float16* __restrict__ w0t, const float* __restrict__ bias0,
    const _Float16* __restrict__ w1t, const float* __restrict__ bias1,
    const _Float16* __restrict__ w2t, const float* __restrict__ bias2,
    float* __restrict__ Y, int out_dim)
{
  __shared__ float lds[4][2][16 * LDH];
  const int wave = threadIdx.x >> 5;
  const int lane = threadIdx.x & 31;
  const int tile = blockIdx.x * 4 + wave;
  const int row0 = tile * 16;
  if (row0 >= n) return;                 // wave-uniform exit; EXEC stays full

  float* H1 = lds[wave][0];
  float* H2 = lds[wave][1];
  const int r  = lane & 15;              // A-frag row / C-frag column
  const int kh = (lane >> 4) * 8;        // K sub-slab select (A/B 16-bit layout)
  const int mb = (lane >> 4) * 8;        // C-frag row base

  // ---- layer 1: X(16x32) @ W0(32x64), K = one 32-slab ----
  v16h a;
  {
    int arow = row0 + r; if (arow >= n) arow = n - 1;   // clamp, keep EXEC full
    const float* xr = X + (size_t)arow * NORB;
    #pragma unroll
    for (int j = 0; j < 8; ++j) a[j]     = (_Float16)xr[kh + j];
    #pragma unroll
    for (int j = 0; j < 8; ++j) a[8 + j] = (_Float16)xr[16 + kh + j];
  }
  #pragma unroll
  for (int nt = 0; nt < 4; ++nt) {
    const int c = nt * 16 + r;
    const _Float16* wc = w0t + (size_t)c * 32;
    v16h b = combine16(*(const v8h*)(wc + kh), *(const v8h*)(wc + 16 + kh));
    v8f acc = {};
    acc = wmma_f16(a, b, acc);
    const float bn = bias0[c];
    #pragma unroll
    for (int j = 0; j < 8; ++j) H1[(mb + j) * LDH + c] = silu(acc[j] + bn);
  }
  asm volatile("s_wait_dscnt 0" ::: "memory");

  // ---- layer 2: H1(16x64) @ W1(64x64), K = two 32-slabs ----
  #pragma unroll
  for (int nt = 0; nt < 4; ++nt) {
    const int c = nt * 16 + r;
    v8f acc = {};
    #pragma unroll
    for (int ks = 0; ks < 2; ++ks) {
      const int kb = ks * 32;
      const float* hr = H1 + r * LDH + kb;
      v16h a2;
      #pragma unroll
      for (int j = 0; j < 8; ++j) a2[j]     = (_Float16)hr[kh + j];
      #pragma unroll
      for (int j = 0; j < 8; ++j) a2[8 + j] = (_Float16)hr[16 + kh + j];
      const _Float16* wc = w1t + (size_t)c * 64 + kb;
      v16h b = combine16(*(const v8h*)(wc + kh), *(const v8h*)(wc + 16 + kh));
      acc = wmma_f16(a2, b, acc);
    }
    const float bn = bias1[c];
    #pragma unroll
    for (int j = 0; j < 8; ++j) H2[(mb + j) * LDH + c] = silu(acc[j] + bn);
  }
  asm volatile("s_wait_dscnt 0" ::: "memory");

  // ---- layer 3: H2(16x64) @ W2t(16pad x 64), N padded with zero columns ----
  {
    const int c = r;
    v8f acc = {};
    #pragma unroll
    for (int ks = 0; ks < 2; ++ks) {
      const int kb = ks * 32;
      const float* hr = H2 + r * LDH + kb;
      v16h a3;
      #pragma unroll
      for (int j = 0; j < 8; ++j) a3[j]     = (_Float16)hr[kh + j];
      #pragma unroll
      for (int j = 0; j < 8; ++j) a3[8 + j] = (_Float16)hr[16 + kh + j];
      const _Float16* wc = w2t + (size_t)c * 64 + kb;
      v16h b = combine16(*(const v8h*)(wc + kh), *(const v8h*)(wc + 16 + kh));
      acc = wmma_f16(a3, b, acc);
    }
    if (c < out_dim) {
      const float bn = bias2[c];
      #pragma unroll
      for (int j = 0; j < 8; ++j) {
        int rr = row0 + mb + j;
        if (rr < n) Y[(size_t)rr * out_dim + c] = acc[j] + bn;
      }
    }
  }
}

// ---------------------------------------------------------------------------
// Per-edge iteration: co_new[nl0] += ic[nl1]*orbital + co_old[nl1]*cut
// ---------------------------------------------------------------------------
__global__ void edge_iter_kernel(const int* __restrict__ nl,
                                 const float* __restrict__ orbital,
                                 const float* __restrict__ cutbuf,
                                 const float* __restrict__ ic,
                                 const float* __restrict__ co_old,
                                 float* __restrict__ co_new, int E)
{
  int e = blockIdx.x * blockDim.x + threadIdx.x;
  if (e >= E) return;
  int i = nl[e];
  int j = nl[E + e];
  float c = cutbuf[e];
  const float* orb = orbital + (size_t)e * 72;
  const float* coj = co_old + (size_t)j * 72;
  const float* icj = ic + (size_t)j * NWAVE_;
  float icv[NWAVE_];
  #pragma unroll
  for (int w = 0; w < NWAVE_; ++w) icv[w] = icj[w];
  float* coi = co_new + (size_t)i * 72;
  #pragma unroll
  for (int a = 0; a < NANG_; ++a)
    #pragma unroll
    for (int w = 0; w < NWAVE_; ++w) {
      float wo = icv[w] * orb[a * NWAVE_ + w] + coj[a * NWAVE_ + w] * c;
      atomicAdd(&coi[a * NWAVE_ + w], wo);
    }
}

// ---------------------------------------------------------------------------
// out = sum_n atom_out[n] * center_factor[n]   (wave reduce + one atomic/wave)
// ---------------------------------------------------------------------------
__global__ void reduce_kernel(const float* __restrict__ atom_out,
                              const float* __restrict__ cf, int n,
                              float* __restrict__ out)
{
  int i = blockIdx.x * blockDim.x + threadIdx.x;
  float v = (i < n) ? atom_out[i] * cf[i] : 0.0f;
  #pragma unroll
  for (int off = 16; off > 0; off >>= 1) v += __shfl_down(v, off, 32);
  if ((threadIdx.x & 31) == 0) atomicAdd(out, v);
}

// ---------------------------------------------------------------------------
extern "C" void kernel_launch(void* const* d_in, const int* in_sizes, int n_in,
                              void* d_out, int out_size, void* d_ws, size_t ws_size,
                              hipStream_t stream)
{
  (void)n_in; (void)out_size; (void)ws_size;
  const float* cart          = (const float*)d_in[0];
  const int*   nl            = (const int*)d_in[1];
  const float* shifts        = (const float*)d_in[2];
  const float* center_factor = (const float*)d_in[3];
  const float* neigh_factor  = (const float*)d_in[4];
  const float* species       = (const float*)d_in[5];
  const float* contracted    = (const float*)d_in[6];
  // params flattened in jax pytree order: emb(3 layers W,b), iters[0..2], out
  const float* ew0 = (const float*)d_in[7];
  const float* eb0 = (const float*)d_in[8];
  const float* ew1 = (const float*)d_in[9];
  const float* eb1 = (const float*)d_in[10];
  const float* ew2 = (const float*)d_in[11];
  const float* eb2 = (const float*)d_in[12];

  const int N = in_sizes[3];   // atoms
  const int E = in_sizes[4];   // edges

  // workspace carve-up (256B aligned)
  char* ws = (char*)d_ws;
  size_t off = 0;
  auto carve = [&](size_t bytes) -> char* {
    char* p = ws + off;
    off = (off + bytes + 255) & ~(size_t)255;
    return p;
  };
  float* orbital  = (float*)carve((size_t)E * 72 * sizeof(float));
  float* cutbuf   = (float*)carve((size_t)E * sizeof(float));
  float* cc       = (float*)carve((size_t)N * EMB * sizeof(float));
  float* co_a     = (float*)carve((size_t)N * 72 * sizeof(float));
  float* co_b     = (float*)carve((size_t)N * 72 * sizeof(float));
  float* density  = (float*)carve((size_t)N * NORB * sizeof(float));
  float* ic       = (float*)carve((size_t)N * NWAVE_ * sizeof(float));
  float* atom_out = (float*)carve((size_t)N * sizeof(float));
  // packed f16 weights: [mlp 0..2 = iters, 3 = out][layer 0..2]
  _Float16* wt0[4]; _Float16* wt1[4]; _Float16* wt2[4];
  for (int m = 0; m < 4; ++m) {
    wt0[m] = (_Float16*)carve((size_t)64 * 32 * sizeof(_Float16));
    wt1[m] = (_Float16*)carve((size_t)64 * 64 * sizeof(_Float16));
    wt2[m] = (_Float16*)carve((size_t)16 * 64 * sizeof(_Float16));
  }

  hipMemsetAsync(co_a, 0, (size_t)N * 72 * sizeof(float), stream);
  hipMemsetAsync(d_out, 0, sizeof(float), stream);

  const int TB = 256;

  // pre-pack all MLP weights to transposed/padded f16 (tiny, once per launch)
  for (int m = 0; m < 4; ++m) {
    int p = (m < 3) ? (13 + 6 * m) : 31;
    int od = (m < 3) ? NWAVE_ : 1;
    pack_w_kernel<<<(64 * 32 + TB - 1) / TB, TB, 0, stream>>>(
        (const float*)d_in[p + 0], 32, HID, 64, wt0[m]);
    pack_w_kernel<<<(64 * 64 + TB - 1) / TB, TB, 0, stream>>>(
        (const float*)d_in[p + 2], 64, HID, 64, wt1[m]);
    pack_w_kernel<<<(16 * 64 + TB - 1) / TB, TB, 0, stream>>>(
        (const float*)d_in[p + 4], 64, od, 16, wt2[m]);
  }

  emb_kernel<<<(N + TB - 1) / TB, TB, 0, stream>>>(species, N, ew0, eb0, ew1, eb1,
                                                   ew2, eb2, cc);
  edge_kernel<<<(E + TB - 1) / TB, TB, 0, stream>>>(nl, cart, shifts, neigh_factor,
                                                    cc, E, orbital, cutbuf, co_a);
  contract_kernel<<<((size_t)N * NORB + TB - 1) / TB, TB, 0, stream>>>(
      co_a, contracted, density, N, 0);

  const int ntiles = (N + 15) / 16;
  const int mlp_blocks = (ntiles + 3) / 4;

  for (int t = 0; t < 3; ++t) {
    const float* ib0 = (const float*)d_in[13 + 6 * t + 1];
    const float* ib1 = (const float*)d_in[13 + 6 * t + 3];
    const float* ib2 = (const float*)d_in[13 + 6 * t + 5];
    mlp_kernel<<<mlp_blocks, 128, 0, stream>>>(density, N, wt0[t], ib0, wt1[t], ib1,
                                               wt2[t], ib2, ic, NWAVE_);
    hipMemcpyAsync(co_b, co_a, (size_t)N * 72 * sizeof(float),
                   hipMemcpyDeviceToDevice, stream);
    edge_iter_kernel<<<(E + TB - 1) / TB, TB, 0, stream>>>(nl, orbital, cutbuf, ic,
                                                           co_a, co_b, E);
    contract_kernel<<<((size_t)N * NORB + TB - 1) / TB, TB, 0, stream>>>(
        co_b, contracted, density, N, 1);
    float* tmp = co_a; co_a = co_b; co_b = tmp;
  }

  const float* ob0 = (const float*)d_in[32];
  const float* ob1 = (const float*)d_in[34];
  const float* ob2 = (const float*)d_in[36];
  mlp_kernel<<<mlp_blocks, 128, 0, stream>>>(density, N, wt0[3], ob0, wt1[3], ob1,
                                             wt2[3], ob2, atom_out, 1);
  reduce_kernel<<<(N + TB - 1) / TB, TB, 0, stream>>>(atom_out, center_factor, N,
                                                      (float*)d_out);
}
